// MDTA_45878840656332
// MI455X (gfx1250) — matrix operand order repaired and checked by reading, not compiled, not run on hardware
//
#include <hip/hip_runtime.h>
#include <math.h>

// ---------------------------------------------------------------------------
// MDTA (Restormer transposed channel attention) for gfx1250 / MI455X.
// Pipeline:
//  K1 (zero):     clear gram + sumsq accumulators in workspace
//  K2 (qkv_dw):   per 16x16 pixel tile: conv1x1 (WMMA bf16) into LDS with halo,
//                 depthwise 3x3, write v (bf16) to ws, wave-reduced sumsq(q,k)
//                 atomics, partial Gram = q k^T via WMMA + f32 atomics.
//  K3 (combine):  M[b] = proj_w @ softmax(normalize(Gram))   (tiny, VALU)
//  K4 (out):      out[b] = M[b] @ v[b]  via WMMA bf16 (K=48), f32 output.
// ---------------------------------------------------------------------------

typedef __bf16 v16bf __attribute__((ext_vector_type(16)));
typedef __bf16 v8bf  __attribute__((ext_vector_type(8)));
typedef float  v8f   __attribute__((ext_vector_type(8)));

union BF16x16 { v16bf v; v8bf h[2]; };

// WMMA D = A(16x32 bf16) x B(32x16 bf16) + C(f32)
__device__ __forceinline__ v8f wmma_bf16(const BF16x16& A, const BF16x16& B, v8f C) {
  return __builtin_amdgcn_wmma_f32_16x16x32_bf16(false, A.v, false, B.v,
                                                 (short)0, C, false, false);
}

// ---- LDS geometry for kernel 2 (all strides keep 16B alignment and avoid
// ---- 64-bank conflicts: stride mod 64 dwords != 0) ----
#define WSTR 72                        // qkv weight row stride (bf16 elems), 144B
#define XSTR 72                        // x tile row stride [px][ch]
#define YSTR 340                       // conv1x1 out stride [ch][px] (f32)
#define DSTR 264                       // dw q/k stride [ch][px] (bf16), 528B
#define HPAD 336                       // 324 halo pixels padded to 21 N-tiles

#define OFF_W 0
#define OFF_X (144 * WSTR * 2)                 // 20736
#define OFF_Y (OFF_X + HPAD * XSTR * 2)        // 69120
#define SMEM_A (OFF_Y + 144 * YSTR * 4)        // 264960 bytes (< 320KB WGP LDS)
#define OFF_DW 0                               // overlays W+X after phase 2

__global__ void __launch_bounds__(256) mdta_zero_kernel(float* p, int n) {
  int i = blockIdx.x * 256 + threadIdx.x;
  if (i < n) p[i] = 0.f;
}

__global__ void __launch_bounds__(256)
mdta_qkv_dw_kernel(const float* __restrict__ x, const float* __restrict__ qkv_w,
                   const float* __restrict__ dw_w, float* __restrict__ gram,
                   float* __restrict__ ssq, __bf16* __restrict__ vbuf) {
  extern __shared__ char smem[];
  __bf16* wlds  = (__bf16*)(smem + OFF_W);
  __bf16* xlds  = (__bf16*)(smem + OFF_X);
  float*  ylds  = (float*)(smem + OFF_Y);
  __bf16* dwlds = (__bf16*)(smem + OFF_DW);

  const int tid = threadIdx.x;
  const int lane = tid & 31, wave = tid >> 5;
  const int b = blockIdx.y;
  const int tx = blockIdx.x & 15, ty = blockIdx.x >> 4;

  // phase 0: zero x region (covers K-pad ch 48..63 and N-pad rows), load W bf16
  {
    unsigned* xz = (unsigned*)xlds;
    for (int i = tid; i < HPAD * XSTR / 2; i += 256) xz[i] = 0u;
    for (int e = tid; e < 144 * 64; e += 256) {
      int oc = e >> 6, ic = e & 63;
      wlds[oc * WSTR + ic] = (ic < 48) ? (__bf16)qkv_w[oc * 48 + ic] : (__bf16)0.f;
    }
  }
  __syncthreads();

  // phase 1: stage 18x18 halo of x into LDS as [px][ch] bf16 (zero outside image)
  {
    const int y0 = ty * 16 - 1, x0 = tx * 16 - 1;
    const float* xb = x + ((size_t)b * 48 << 16);
    for (int e = tid; e < 324 * 48; e += 256) {
      int hp = e % 324, c = e / 324;
      int hr = hp / 18, hc = hp % 18;
      int gy = y0 + hr, gx = x0 + hc;
      float v = 0.f;
      if ((unsigned)gy < 256u && (unsigned)gx < 256u)
        v = xb[((size_t)c << 16) + gy * 256 + gx];
      xlds[hp * XSTR + c] = (__bf16)v;
    }
  }
  __syncthreads();

  // phase 2: conv1x1 GEMM  y[144][324] = W[144x48] * x[48 x 324]  (WMMA bf16)
  {
    const int m15 = lane & 15;
    const int klo = (lane >= 16) ? 8 : 0;
    const int kb  = (lane >= 16) ? 16 : 0;
    for (int job = wave; job < 9 * 21; job += 8) {
      int mt = job / 21, nt = job % 21;
      v8f acc = {};
#pragma unroll
      for (int kt = 0; kt < 2; ++kt) {   // K = 48 padded to 64
        BF16x16 A, B;
        const __bf16* ar = wlds + (mt * 16 + m15) * WSTR + kt * 32;
        A.h[0] = *(const v8bf*)(ar + klo);
        A.h[1] = *(const v8bf*)(ar + klo + 16);
        const __bf16* br = xlds + (nt * 16 + m15) * XSTR + kt * 32;
        B.h[0] = *(const v8bf*)(br + klo);
        B.h[1] = *(const v8bf*)(br + klo + 16);
        acc = wmma_bf16(A, B, acc);
      }
      int col = nt * 16 + m15;
      int rbase = mt * 16 + ((lane >= 16) ? 8 : 0);
#pragma unroll
      for (int r = 0; r < 8; ++r) ylds[(rbase + r) * YSTR + col] = acc[r];
    }
  }
  __syncthreads();

  // phase 3: depthwise 3x3 over LDS halo; q/k -> LDS bf16, v -> global bf16
  {
    const int i = tid >> 4, j = tid & 15;
    const int gp = (ty * 16 + i) * 256 + tx * 16 + j;
    __bf16* vb = vbuf + ((size_t)b * 48 << 16) + gp;
    for (int c = 0; c < 144; ++c) {
      const float* wd = dw_w + c * 9;           // uniform -> scalar loads
      const float* yc = ylds + c * YSTR + i * 18 + j;
      float s = 0.f;
#pragma unroll
      for (int dy = 0; dy < 3; ++dy)
#pragma unroll
        for (int dx = 0; dx < 3; ++dx)
          s = fmaf(wd[dy * 3 + dx], yc[dy * 18 + dx], s);
      if (c < 96) dwlds[c * DSTR + tid] = (__bf16)s;
      else        vb[(size_t)(c - 96) << 16] = (__bf16)s;
    }
  }
  __syncthreads();

  // phase 4: per-channel sum-of-squares for L2 normalization (wave reduce)
  for (int c = wave; c < 96; c += 8) {
    float s = 0.f;
    for (int p = lane; p < 256; p += 32) {
      float v = (float)dwlds[c * DSTR + p];
      s = fmaf(v, v, s);
    }
#pragma unroll
    for (int off = 16; off >= 1; off >>= 1) s += __shfl_xor(s, off, 32);
    if (lane == 0) atomicAdd(&ssq[b * 96 + c], s);
  }

  // phase 5: partial Gram[48x48] = q * k^T over this tile's 256 pixels (WMMA)
  {
    const int m15 = lane & 15;
    const int klo = (lane >= 16) ? 8 : 0;
    const int kb  = (lane >= 16) ? 16 : 0;
    for (int job = wave; job < 9; job += 8) {
      int mt = job / 3, nt = job % 3;
      v8f acc = {};
#pragma unroll
      for (int kt = 0; kt < 8; ++kt) {   // K = 256 pixels
        BF16x16 A, B;
        const __bf16* ar = dwlds + (mt * 16 + m15) * DSTR + kt * 32;
        A.h[0] = *(const v8bf*)(ar + klo);
        A.h[1] = *(const v8bf*)(ar + klo + 16);
        const __bf16* br = dwlds + (48 + nt * 16 + m15) * DSTR + kt * 32;
        B.h[0] = *(const v8bf*)(br + klo);
        B.h[1] = *(const v8bf*)(br + klo + 16);
        acc = wmma_bf16(A, B, acc);
      }
      int n = nt * 16 + m15;
      int mbase = mt * 16 + ((lane >= 16) ? 8 : 0);
#pragma unroll
      for (int r = 0; r < 8; ++r)
        atomicAdd(&gram[(b * 48 + mbase + r) * 48 + n], acc[r]);
    }
  }
}

// K3: M[b] = proj_w @ softmax( Gram / (|q_c||k_d|) * temperature )
__global__ void __launch_bounds__(64)
mdta_combine_kernel(const float* __restrict__ gram, const float* __restrict__ ssq,
                    const float* __restrict__ proj_w, const float* __restrict__ temp,
                    float* __restrict__ mmat) {
  __shared__ float attn[48 * 48];
  __shared__ float nq[48], nk[48];
  const int b = blockIdx.x, tid = threadIdx.x;
  if (tid < 48) {
    nq[tid] = fmaxf(sqrtf(ssq[b * 96 + tid]), 1e-12f);
    nk[tid] = fmaxf(sqrtf(ssq[b * 96 + 48 + tid]), 1e-12f);
  }
  __syncthreads();
  if (tid < 48) {
    const int c = tid;
    const float t = temp[0];
    float l[48];
    float m = -3.4e38f;
#pragma unroll
    for (int d = 0; d < 48; ++d) {
      float v = gram[(b * 48 + c) * 48 + d] / (nq[c] * nk[d]) * t;
      l[d] = v; m = fmaxf(m, v);
    }
    float ssum = 0.f;
#pragma unroll
    for (int d = 0; d < 48; ++d) { float e = __expf(l[d] - m); l[d] = e; ssum += e; }
    float inv = 1.f / ssum;
#pragma unroll
    for (int d = 0; d < 48; ++d) attn[c * 48 + d] = l[d] * inv;
  }
  __syncthreads();
  if (tid < 48) {
    const int i = tid;
    for (int j = 0; j < 48; ++j) {
      float s = 0.f;
#pragma unroll
      for (int d = 0; d < 48; ++d) s = fmaf(proj_w[i * 48 + d], attn[d * 48 + j], s);
      mmat[(b * 48 + i) * 48 + j] = s;
    }
  }
}

// K4: out[b][48][N] = M[b][48x48] @ v[b][48][N]   (WMMA bf16, K=48 padded 64)
#define C_OFF_A 0
#define C_OFF_V (48 * 72 * 2)                  // 6912
#define SMEM_C (C_OFF_V + 1024 * 72 * 2)       // 154368 bytes

__global__ void __launch_bounds__(256)
mdta_out_kernel(const float* __restrict__ mmat, const __bf16* __restrict__ vbuf,
                float* __restrict__ out) {
  extern __shared__ char smem[];
  __bf16* alds = (__bf16*)(smem + C_OFF_A);
  __bf16* vlds = (__bf16*)(smem + C_OFF_V);
  const int tid = threadIdx.x, lane = tid & 31, wave = tid >> 5;
  const int b = blockIdx.y;
  const int px0 = blockIdx.x * 1024;

  for (int e = tid; e < 48 * 64; e += 256) {
    int i = e >> 6, d = e & 63;
    alds[i * 72 + d] = (d < 48) ? (__bf16)mmat[(b * 48 + i) * 48 + d] : (__bf16)0.f;
  }
  const __bf16* vb = vbuf + ((size_t)b * 48 << 16) + px0;
  for (int e = tid; e < 48 * 1024; e += 256) {   // coalesced read, [px][ch] in LDS
    int px = e & 1023, ch = e >> 10;
    vlds[px * 72 + ch] = vb[((size_t)ch << 16) + px];
  }
  for (int e = tid; e < 16 * 1024; e += 256) {   // zero K-pad ch 48..63
    int px = e & 1023, ch = 48 + (e >> 10);
    vlds[px * 72 + ch] = (__bf16)0.f;
  }
  __syncthreads();

  const int m15 = lane & 15;
  const int klo = (lane >= 16) ? 8 : 0;
  const int kb  = (lane >= 16) ? 16 : 0;
  float* ob = out + ((size_t)b * 48 << 16) + px0;
  for (int job = wave; job < 3 * 64; job += 8) {
    int mt = job % 3, nt = job / 3;
    v8f acc = {};
#pragma unroll
    for (int kt = 0; kt < 2; ++kt) {
      BF16x16 A, B;
      const __bf16* ar = alds + (mt * 16 + m15) * 72 + kt * 32;
      A.h[0] = *(const v8bf*)(ar + klo);
      A.h[1] = *(const v8bf*)(ar + klo + 16);
      const __bf16* br = vlds + (nt * 16 + m15) * 72 + kt * 32;
      B.h[0] = *(const v8bf*)(br + klo);
      B.h[1] = *(const v8bf*)(br + klo + 16);
      acc = wmma_bf16(A, B, acc);
    }
    int n = nt * 16 + m15;
    int mbase = mt * 16 + ((lane >= 16) ? 8 : 0);
#pragma unroll
    for (int r = 0; r < 8; ++r)
      ob[((size_t)(mbase + r) << 16) + n] = acc[r];
  }
}

extern "C" void kernel_launch(void* const* d_in, const int* in_sizes, int n_in,
                              void* d_out, int out_size, void* d_ws, size_t ws_size,
                              hipStream_t stream) {
  const float* x      = (const float*)d_in[0];
  const float* qkv_w  = (const float*)d_in[1];
  const float* dw_w   = (const float*)d_in[2];
  const float* proj_w = (const float*)d_in[3];
  const float* temp   = (const float*)d_in[4];
  float* out = (float*)d_out;

  // workspace layout (floats): gram[4*48*48] | ssq[4*96] | mmat[4*48*48] | vbuf bf16
  float* gram = (float*)d_ws;
  float* ssq  = gram + 4 * 48 * 48;
  float* mmat = ssq + 4 * 96;
  __bf16* vbuf = (__bf16*)(mmat + 4 * 48 * 48);   // 4*48*65536 bf16 = 25 MB

  (void)hipFuncSetAttribute((const void*)mdta_qkv_dw_kernel,
                            hipFuncAttributeMaxDynamicSharedMemorySize, SMEM_A);
  (void)hipFuncSetAttribute((const void*)mdta_out_kernel,
                            hipFuncAttributeMaxDynamicSharedMemorySize, SMEM_C);

  const int nacc = 4 * 48 * 48 + 4 * 96;
  mdta_zero_kernel<<<dim3((nacc + 255) / 256), 256, 0, stream>>>((float*)d_ws, nacc);
  mdta_qkv_dw_kernel<<<dim3(256, 4), 256, SMEM_A, stream>>>(x, qkv_w, dw_w,
                                                            gram, ssq, vbuf);
  mdta_combine_kernel<<<dim3(4), 64, 0, stream>>>(gram, ssq, proj_w, temp, mmat);
  mdta_out_kernel<<<dim3(64, 4), 256, SMEM_C, stream>>>(mmat, vbuf, out);
}
